// AdaptConvCls_35656818492090
// MI455X (gfx1250) — matrix operand order repaired and checked by reading, not compiled, not run on hardware
//
#include <hip/hip_runtime.h>
#include <hip/hip_bf16.h>
#include <stdint.h>

typedef __attribute__((ext_vector_type(16))) _Float16 v16h;
typedef __attribute__((ext_vector_type(8)))  float    v8f;

#define NPTS 1024
#define BATCH 2
#define KNNK 20
#define LRELUF 0.2f
#define NP_TOT (BATCH * NPTS * KNNK)   // 40960 (n,k) points

__device__ __forceinline__ float leakyf(float v) { return v >= 0.f ? v : LRELUF * v; }

__device__ __forceinline__ void atomicMaxF(float* addr, float v) {
  __hip_atomic_fetch_max(addr, v, __ATOMIC_RELAXED, __HIP_MEMORY_SCOPE_AGENT);
}

// packed f16 pair store into LDS (dst must be 4B aligned)
__device__ __forceinline__ void st_lds_h2(_Float16* dst, float a, float b) {
  union { _Float16 h[2]; uint32_t u; } t;
  t.h[0] = (_Float16)a; t.h[1] = (_Float16)b;
  *reinterpret_cast<uint32_t*>(dst) = t.u;
}

// load 16 contiguous halves (two b128 loads)
__device__ __forceinline__ v16h ld16h(const _Float16* p) {
  union { uint4 u[2]; v16h h; } t;
  t.u[0] = *reinterpret_cast<const uint4*>(p);
  t.u[1] = *reinterpret_cast<const uint4*>(p + 8);
  return t.h;
}

// A-fragment: halves [g*8 .. g*8+7] and [16+g*8 .. 16+g*8+7] of a K-contiguous row
__device__ __forceinline__ v16h ldAfrag(const _Float16* row, int g) {
  union { uint4 u[2]; v16h h; } t;
  t.u[0] = *reinterpret_cast<const uint4*>(row + g * 8);
  t.u[1] = *reinterpret_cast<const uint4*>(row + 16 + g * 8);
  return t.h;
}

// ---------------------------------------------------------------------------
__global__ void fill_k(float* p, size_t n, float v) {
  size_t i = (size_t)blockIdx.x * blockDim.x + threadIdx.x;
  if (i < n) p[i] = v;
}

// ---------------------------------------------------------------------------
__global__ void norms_k(const float* __restrict__ X, int batchStride, int C,
                        float* __restrict__ xx) {
  int t = blockIdx.x * blockDim.x + threadIdx.x;
  if (t >= BATCH * NPTS) return;
  int b = t / NPTS, n = t % NPTS;
  const float* p = X + (size_t)b * batchStride + n;
  float s = 0.f;
  for (int c = 0; c < C; ++c) { float v = p[(size_t)c * NPTS]; s += v * v; }
  xx[t] = s;
}

// ---------------------------------------------------------------------------
// Gram matrix G[b] = X^T X (N x N). Tiles stored f16 transposed [point][chan].
// block = 128 threads (4 waves): 64x64 output tile, each wave 16 rows.
__global__ __launch_bounds__(128) void gram_k(const float* __restrict__ X,
                                              int batchStride, int C,
                                              float* __restrict__ G) {
  __shared__ _Float16 tI[64][40];   // 32 chans + 8 pad
  __shared__ _Float16 tJ[64][40];
  int b  = blockIdx.z;
  int bi = blockIdx.y * 64, bj = blockIdx.x * 64;
  int tid = threadIdx.x, lane = tid & 31, w = tid >> 5;
  int m16 = lane & 15, g = lane >> 4;

  v8f acc[4] = {};
  for (int cb = 0; cb < C; cb += 32) {
    __syncthreads();
    for (int e = tid; e < 16 * 64; e += 128) {
      int cpair = e >> 6, nn = e & 63;
      int c0 = cb + 2 * cpair;
      float vi0 = 0.f, vi1 = 0.f, vj0 = 0.f, vj1 = 0.f;
      size_t rb = (size_t)b * batchStride;
      if (c0 < C)     { vi0 = X[rb + (size_t)c0 * NPTS + bi + nn];
                        vj0 = X[rb + (size_t)c0 * NPTS + bj + nn]; }
      if (c0 + 1 < C) { vi1 = X[rb + (size_t)(c0 + 1) * NPTS + bi + nn];
                        vj1 = X[rb + (size_t)(c0 + 1) * NPTS + bj + nn]; }
      st_lds_h2(&tI[nn][2 * cpair], vi0, vi1);
      st_lds_h2(&tJ[nn][2 * cpair], vj0, vj1);
    }
    __syncthreads();
    v16h af = ldAfrag(&tI[w * 16 + m16][0], g);
#pragma unroll
    for (int jt = 0; jt < 4; ++jt) {
      v16h bf = ldAfrag(&tJ[jt * 16 + m16][0], g);
      acc[jt] = __builtin_amdgcn_wmma_f32_16x16x32_f16(false, af, false, bf,
                                                       (short)0, acc[jt], false, false);
    }
  }
#pragma unroll
  for (int jt = 0; jt < 4; ++jt)
#pragma unroll
    for (int r = 0; r < 8; ++r) {
      int row = bi + w * 16 + g * 8 + r;
      int col = bj + jt * 16 + m16;
      G[(size_t)b * NPTS * NPTS + (size_t)row * NPTS + col] = acc[jt][r];
    }
}

// ---------------------------------------------------------------------------
__global__ void topk_k(const float* __restrict__ G, const float* __restrict__ xx,
                       int* __restrict__ idx) {
  int t = blockIdx.x * blockDim.x + threadIdx.x;
  if (t >= BATCH * NPTS) return;
  int b = t / NPTS, n = t % NPTS;
  const float* row = G + (size_t)b * NPTS * NPTS + (size_t)n * NPTS;
  const float* xxb = xx + b * NPTS;
  float xn = xxb[n];
  float best[KNNK];
  int   bix[KNNK];
#pragma unroll
  for (int i = 0; i < KNNK; ++i) { best[i] = -3.0e38f; bix[i] = 0; }
  for (int mm = 0; mm < NPTS; ++mm) {
    float d = 2.f * row[mm] - xn - xxb[mm];
    if (d > best[KNNK - 1]) {
      int pos = KNNK - 1;
      while (pos > 0 && best[pos - 1] < d) {
        best[pos] = best[pos - 1]; bix[pos] = bix[pos - 1]; --pos;
      }
      best[pos] = d; bix[pos] = mm;
    }
  }
  int* o = idx + (size_t)t * KNNK;
  for (int i = 0; i < KNNK; ++i) o[i] = bix[i];
}

// ---------------------------------------------------------------------------
// weight prep: W1r[o][c*IN_C+i] = (f16) w1[(o*IN_C+i)][c]
__global__ void prep_w1r_k(const float* __restrict__ w1, int IN_C,
                           _Float16* __restrict__ out) {
  int t = blockIdx.x * blockDim.x + threadIdx.x;
  int kd = 64 * IN_C;
  if (t >= 64 * kd) return;
  int o = t / kd, rem = t % kd;
  int c = rem / IN_C, i = rem % IN_C;
  out[t] = (_Float16)w1[(size_t)(o * IN_C + i) * 64 + c];
}

__global__ void prep_half_k(const float* __restrict__ w, int n, _Float16* __restrict__ out) {
  int t = blockIdx.x * blockDim.x + threadIdx.x;
  if (t < n) out[t] = (_Float16)w[t];
}

// ---------------------------------------------------------------------------
// shared helper: load point descriptors for a 16-point tile
__device__ __forceinline__ void load_tile_pts(const int* __restrict__ idx, int pt0,
                                              int tid, int* ml, int* nl, int* bl) {
  if (tid < 16) {
    int q = pt0 + tid;
    int b = q / (NPTS * KNNK);
    int n = (q / KNNK) % NPTS;
    int k = q % KNNK;
    ml[tid] = idx[(size_t)(b * NPTS + n) * KNNK + k];
    nl[tid] = n; bl[tid] = b;
  }
}

// ---------------------------------------------------------------------------
// AdaptiveConv 1 (IN_C=6, KD=384): precompute full V tile (384x16) in f16 LDS.
__global__ __launch_bounds__(128) void adapt_conv1_k(
    const float* __restrict__ coord, const int* __restrict__ idx,
    const float* __restrict__ w0, const float* __restrict__ g0, const float* __restrict__ b0,
    const _Float16* __restrict__ w1r,
    const float* __restrict__ g1, const float* __restrict__ b1,
    float* __restrict__ feat, int rowOff) {
  constexpr int KD = 384;
  __shared__ float p6s[6][16];
  __shared__ float y0s[64][16];
  __shared__ _Float16 vts[16][KD + 8];   // transposed V tile, f16, padded
  __shared__ int ml[16], nl[16], bl[16];

  int tid = threadIdx.x, lane = tid & 31, w = tid >> 5;
  int pt0 = blockIdx.x * 16;
  load_tile_pts(idx, pt0, tid, ml, nl, bl);
  __syncthreads();
  // graph feature of coord = both p and xf (CSRC=3, IN_C=6): p6s
  for (int e = tid; e < 6 * 16; e += 128) {
    int r = e >> 4, p = e & 15;
    size_t base = (size_t)bl[p] * 3 * NPTS;
    float v = (r < 3) ? coord[base + r * NPTS + ml[p]] - coord[base + r * NPTS + nl[p]]
                      : coord[base + (r - 3) * NPTS + nl[p]];
    p6s[r][p] = v;
  }
  __syncthreads();
  for (int e = tid; e < 64 * 16; e += 128) {
    int c = e >> 4, p = e & 15;
    float s = 0.f;
#pragma unroll
    for (int j = 0; j < 6; ++j) s += w0[c * 6 + j] * p6s[j][p];
    y0s[c][p] = leakyf(s * g0[c] + b0[c]);
  }
  __syncthreads();
  // V[p][c*6+i] = y0[c][p] * p6s[i][p]  (xf == p6 for stage 1)
  for (int e = tid; e < (KD / 2) * 16; e += 128) {
    int kp = e >> 4, p = e & 15;
    int k0 = 2 * kp, k1 = k0 + 1;
    float v0 = y0s[k0 / 6][p] * p6s[k0 % 6][p];
    float v1 = y0s[k1 / 6][p] * p6s[k1 % 6][p];
    st_lds_h2(&vts[p][k0], v0, v1);
  }
  __syncthreads();

  int m16 = lane & 15, g = lane >> 4;
  int o0 = w * 16;
  v8f acc = {};
  const _Float16* arow = w1r + (size_t)(o0 + m16) * KD;
#pragma unroll
  for (int kb = 0; kb < KD; kb += 32) {
    v16h af = ldAfrag(arow + kb, g);
    v16h bf = ldAfrag(&vts[m16][kb], g);
    acc = __builtin_amdgcn_wmma_f32_16x16x32_f16(false, af, false, bf,
                                                 (short)0, acc, false, false);
  }
#pragma unroll
  for (int r = 0; r < 8; ++r) {
    int ch = o0 + g * 8 + r;
    float v = leakyf(acc[r] * g1[ch] + b1[ch]);
    atomicMaxF(feat + (size_t)bl[m16] * 512 * NPTS + (size_t)(rowOff + ch) * NPTS + nl[m16], v);
  }
}

// ---------------------------------------------------------------------------
// AdaptiveConv 2 (IN_C=128, KD=8192): c uniform per 32-chunk -> V chunk is
// packed-f16 xf * splat(y0[c,p]) via v_pk_mul_f16.
__global__ __launch_bounds__(128) void adapt_conv2_k(
    const float* __restrict__ coord,
    const float* __restrict__ xsrc, int xsrcBatchStride,   // (B,64,N)
    const int* __restrict__ idx,
    const float* __restrict__ w0, const float* __restrict__ g0, const float* __restrict__ b0,
    const _Float16* __restrict__ w1r,
    const float* __restrict__ g1, const float* __restrict__ b1,
    float* __restrict__ feat, int rowOff) {
  constexpr int KD = 8192;
  __shared__ float p6s[6][16];
  __shared__ float y0s[64][16];
  __shared__ _Float16 xfs[16][136];      // transposed xf tile (128 + 8 pad), f16
  __shared__ int ml[16], nl[16], bl[16];

  int tid = threadIdx.x, lane = tid & 31, w = tid >> 5;
  int pt0 = blockIdx.x * 16;
  load_tile_pts(idx, pt0, tid, ml, nl, bl);
  __syncthreads();
  for (int e = tid; e < 6 * 16; e += 128) {
    int r = e >> 4, p = e & 15;
    size_t base = (size_t)bl[p] * 3 * NPTS;
    float v = (r < 3) ? coord[base + r * NPTS + ml[p]] - coord[base + r * NPTS + nl[p]]
                      : coord[base + (r - 3) * NPTS + nl[p]];
    p6s[r][p] = v;
  }
  // xf (128 rows): rows [0,64) = neigh-center, [64,128) = center
  for (int e = tid; e < 64 * 16; e += 128) {
    int rp = e >> 4, p = e & 15;
    const float* base = xsrc + (size_t)bl[p] * xsrcBatchStride;
    int r0 = 2 * rp, r1 = r0 + 1;
    float v0 = (r0 < 64) ? base[(size_t)r0 * NPTS + ml[p]] - base[(size_t)r0 * NPTS + nl[p]]
                         : base[(size_t)(r0 - 64) * NPTS + nl[p]];
    float v1 = (r1 < 64) ? base[(size_t)r1 * NPTS + ml[p]] - base[(size_t)r1 * NPTS + nl[p]]
                         : base[(size_t)(r1 - 64) * NPTS + nl[p]];
    st_lds_h2(&xfs[p][r0], v0, v1);
  }
  __syncthreads();
  for (int e = tid; e < 64 * 16; e += 128) {
    int c = e >> 4, p = e & 15;
    float s = 0.f;
#pragma unroll
    for (int j = 0; j < 6; ++j) s += w0[c * 6 + j] * p6s[j][p];
    y0s[c][p] = leakyf(s * g0[c] + b0[c]);
  }
  __syncthreads();

  int m16 = lane & 15, g = lane >> 4;
  int o0 = w * 16;
  v8f acc = {};
  const _Float16* arow = w1r + (size_t)(o0 + m16) * KD;
  for (int c = 0; c < 64; ++c) {
    _Float16 yh = (_Float16)y0s[c][m16];
    v16h yv;
#pragma unroll
    for (int e = 0; e < 16; ++e) yv[e] = yh;
#pragma unroll
    for (int s = 0; s < 4; ++s) {
      int kb = c * 128 + s * 32;
      v16h af = ldAfrag(arow + kb, g);
      __builtin_prefetch(arow + kb + 128, 0, 1);
      v16h xh = ldAfrag(&xfs[m16][s * 32], g);
      v16h bf = xh * yv;               // v_pk_mul_f16 x8
      acc = __builtin_amdgcn_wmma_f32_16x16x32_f16(false, af, false, bf,
                                                   (short)0, acc, false, false);
    }
  }
#pragma unroll
  for (int r = 0; r < 8; ++r) {
    int ch = o0 + g * 8 + r;
    float v = leakyf(acc[r] * g1[ch] + b1[ch]);
    atomicMaxF(feat + (size_t)bl[m16] * 512 * NPTS + (size_t)(rowOff + ch) * NPTS + nl[m16], v);
  }
}

// ---------------------------------------------------------------------------
// Edge-conv MLP: D(Ox16) = W(Ox2C) @ GF(2Cx16); GF stored f16 transposed.
__global__ __launch_bounds__(128) void graph_mlp_k(
    const float* __restrict__ xsrc, int xsrcBatchStride, int CSRC,
    const int* __restrict__ idx,
    const _Float16* __restrict__ W, const float* __restrict__ gg, const float* __restrict__ bb,
    int O, float* __restrict__ feat, int rowOff) {
  __shared__ _Float16 gfs[16][264];     // up to 256 chans + 8 pad
  __shared__ int ml[16], nl[16], bl[16];
  int tid = threadIdx.x, lane = tid & 31, w = tid >> 5;
  int C2 = 2 * CSRC;
  int pt0 = blockIdx.x * 16;
  load_tile_pts(idx, pt0, tid, ml, nl, bl);
  __syncthreads();
  for (int e = tid; e < (C2 >> 1) * 16; e += 128) {
    int rp = e >> 4, p = e & 15;
    const float* base = xsrc + (size_t)bl[p] * xsrcBatchStride;
    int r0 = 2 * rp, r1 = r0 + 1;
    float v0 = (r0 < CSRC) ? base[(size_t)r0 * NPTS + ml[p]] - base[(size_t)r0 * NPTS + nl[p]]
                           : base[(size_t)(r0 - CSRC) * NPTS + nl[p]];
    float v1 = (r1 < CSRC) ? base[(size_t)r1 * NPTS + ml[p]] - base[(size_t)r1 * NPTS + nl[p]]
                           : base[(size_t)(r1 - CSRC) * NPTS + nl[p]];
    st_lds_h2(&gfs[p][r0], v0, v1);
  }
  __syncthreads();

  int m16 = lane & 15, g = lane >> 4;
  for (int ot = w; ot < O / 16; ot += 4) {
    int o0 = ot * 16;
    v8f acc = {};
    const _Float16* arow = W + (size_t)(o0 + m16) * C2;
    for (int kb = 0; kb < C2; kb += 32) {
      v16h af = ldAfrag(arow + kb, g);
      v16h bf = ldAfrag(&gfs[m16][kb], g);
      acc = __builtin_amdgcn_wmma_f32_16x16x32_f16(false, af, false, bf,
                                                   (short)0, acc, false, false);
    }
#pragma unroll
    for (int r = 0; r < 8; ++r) {
      int ch = o0 + g * 8 + r;
      float v = leakyf(acc[r] * gg[ch] + bb[ch]);
      atomicMaxF(feat + (size_t)bl[m16] * 512 * NPTS + (size_t)(rowOff + ch) * NPTS + nl[m16], v);
    }
  }
}

// ---------------------------------------------------------------------------
// c5: h5(1024 x B*N) = leaky(bn(W5(1024x512) @ feat(512 x B*N)))
__global__ __launch_bounds__(128) void conv1d_c5_k(
    const float* __restrict__ feat, const _Float16* __restrict__ W,
    const float* __restrict__ gg, const float* __restrict__ bb,
    float* __restrict__ h5) {
  __shared__ _Float16 fs[16][520];      // 512 chans + 8 pad, transposed, f16
  int tid = threadIdx.x, lane = tid & 31, w = tid >> 5;
  int pt0 = blockIdx.x * 16;
  int b = pt0 / NPTS, n0 = pt0 % NPTS;
  for (int e = tid; e < 256 * 16; e += 128) {
    int nn = e & 15, cpair = e >> 4;
    int c0 = 2 * cpair;
    size_t base = (size_t)b * 512 * NPTS + n0 + nn;
    float v0 = feat[base + (size_t)c0 * NPTS];
    float v1 = feat[base + (size_t)(c0 + 1) * NPTS];
    st_lds_h2(&fs[nn][c0], v0, v1);
  }
  __syncthreads();
  int m16 = lane & 15, g = lane >> 4;
  for (int ot = w; ot < 64; ot += 4) {
    int o0 = ot * 16;
    v8f acc = {};
    const _Float16* arow = W + (size_t)(o0 + m16) * 512;
#pragma unroll 4
    for (int kb = 0; kb < 512; kb += 32) {
      v16h af = ldAfrag(arow + kb, g);
      __builtin_prefetch(arow + kb + 128, 0, 1);
      v16h bf = ldAfrag(&fs[m16][kb], g);
      acc = __builtin_amdgcn_wmma_f32_16x16x32_f16(false, af, false, bf,
                                                   (short)0, acc, false, false);
    }
#pragma unroll
    for (int r = 0; r < 8; ++r) {
      int ch = o0 + g * 8 + r;
      float v = leakyf(acc[r] * gg[ch] + bb[ch]);
      h5[(size_t)b * 1024 * NPTS + (size_t)ch * NPTS + n0 + m16] = v;
    }
  }
}

// ---------------------------------------------------------------------------
__global__ void pool_k(const float* __restrict__ h5, float* __restrict__ pooled) {
  int tid = threadIdx.x, lane = tid & 31, w = tid >> 5;
  int row = blockIdx.x * 4 + w;        // b*1024 + ch
  if (row >= BATCH * 1024) return;
  int b = row >> 10, ch = row & 1023;
  const float* p = h5 + (size_t)row * NPTS;
  float mx = -3.0e38f, sm = 0.f;
  for (int i = lane; i < NPTS; i += 32) { float v = p[i]; mx = fmaxf(mx, v); sm += v; }
#pragma unroll
  for (int o = 16; o > 0; o >>= 1) {
    mx = fmaxf(mx, __shfl_xor(mx, o, 32));
    sm += __shfl_xor(sm, o, 32);
  }
  if (lane == 0) {
    pooled[b * 2048 + ch] = mx;
    pooled[b * 2048 + 1024 + ch] = sm * (1.f / NPTS);
  }
}

// ---------------------------------------------------------------------------
__global__ void fc_k(const float* __restrict__ in, int CIN, const float* __restrict__ W,
                     const float* __restrict__ bias, const float* __restrict__ gg,
                     const float* __restrict__ bb, int relu, int O, float* __restrict__ out) {
  int tid = threadIdx.x, lane = tid & 31, w = tid >> 5;
  int t = blockIdx.x * 4 + w;          // b*O + o
  if (t >= BATCH * O) return;
  int b = t / O, o = t % O;
  const float* x  = in + (size_t)b * CIN;
  const float* wr = W + (size_t)o * CIN;
  float s = 0.f;
  for (int i = lane; i < CIN; i += 32) s += x[i] * wr[i];
#pragma unroll
  for (int off = 16; off > 0; off >>= 1) s += __shfl_xor(s, off, 32);
  if (lane == 0) {
    if (bias) s += bias[o];
    if (gg) s = s * gg[o] + bb[o];
    if (relu) s = leakyf(s);
    out[(size_t)b * O + o] = s;
  }
}

// ---------------------------------------------------------------------------
extern "C" void kernel_launch(void* const* d_in, const int* in_sizes, int n_in,
                              void* d_out, int out_size, void* d_ws, size_t ws_size,
                              hipStream_t stream) {
  const float* x = (const float*)d_in[0];
  const float* ac1_w0 = (const float*)d_in[1];
  const float* ac1_g0 = (const float*)d_in[2];
  const float* ac1_b0 = (const float*)d_in[3];
  const float* ac1_w1 = (const float*)d_in[4];
  const float* ac1_g1 = (const float*)d_in[5];
  const float* ac1_b1 = (const float*)d_in[6];
  const float* ac2_w0 = (const float*)d_in[7];
  const float* ac2_g0 = (const float*)d_in[8];
  const float* ac2_b0 = (const float*)d_in[9];
  const float* ac2_w1 = (const float*)d_in[10];
  const float* ac2_g1 = (const float*)d_in[11];
  const float* ac2_b1 = (const float*)d_in[12];
  const float* c3_w = (const float*)d_in[13];
  const float* c3_g = (const float*)d_in[14];
  const float* c3_b = (const float*)d_in[15];
  const float* c4_w = (const float*)d_in[16];
  const float* c4_g = (const float*)d_in[17];
  const float* c4_b = (const float*)d_in[18];
  const float* c5_w = (const float*)d_in[19];
  const float* c5_g = (const float*)d_in[20];
  const float* c5_b = (const float*)d_in[21];
  const float* h1_w = (const float*)d_in[22];
  const float* h1_g = (const float*)d_in[23];
  const float* h1_b = (const float*)d_in[24];
  const float* h2_w = (const float*)d_in[25];
  const float* h2_bias = (const float*)d_in[26];
  const float* h2_g = (const float*)d_in[27];
  const float* h2_b = (const float*)d_in[28];
  const float* h3_w = (const float*)d_in[29];
  const float* h3_bias = (const float*)d_in[30];
  float* out = (float*)d_out;

  char* ws = (char*)d_ws;
  size_t off = 0;
  auto take = [&](size_t bytes) -> char* {
    char* p = ws + off;
    off += (bytes + 255) & ~(size_t)255;
    return p;
  };
  float*    G      = (float*)take((size_t)BATCH * NPTS * NPTS * 4);
  float*    xx     = (float*)take((size_t)BATCH * NPTS * 4);
  int*      idx    = (int*)take((size_t)BATCH * NPTS * KNNK * 4);
  _Float16* w1r1   = (_Float16*)take((size_t)64 * 384 * 2);
  _Float16* w1r2   = (_Float16*)take((size_t)64 * 8192 * 2);
  _Float16* wc3    = (_Float16*)take((size_t)128 * 128 * 2);
  _Float16* wc4    = (_Float16*)take((size_t)256 * 256 * 2);
  _Float16* wc5    = (_Float16*)take((size_t)1024 * 512 * 2);
  float*    feat   = (float*)take((size_t)BATCH * 512 * NPTS * 4);
  float*    h5     = (float*)take((size_t)BATCH * 1024 * NPTS * 4);
  float*    pooled = (float*)take((size_t)BATCH * 2048 * 4);
  float*    fb1    = (float*)take((size_t)BATCH * 512 * 4);
  float*    fb2    = (float*)take((size_t)BATCH * 256 * 4);

  auto cdiv = [](int a, int b) { return (a + b - 1) / b; };

  prep_w1r_k<<<cdiv(64 * 384, 256), 256, 0, stream>>>(ac1_w1, 6, w1r1);
  prep_w1r_k<<<cdiv(64 * 8192, 256), 256, 0, stream>>>(ac2_w1, 128, w1r2);
  prep_half_k<<<cdiv(128 * 128, 256), 256, 0, stream>>>(c3_w, 128 * 128, wc3);
  prep_half_k<<<cdiv(256 * 256, 256), 256, 0, stream>>>(c4_w, 256 * 256, wc4);
  prep_half_k<<<cdiv(1024 * 512, 256), 256, 0, stream>>>(c5_w, 1024 * 512, wc5);
  fill_k<<<cdiv(BATCH * 512 * NPTS, 256), 256, 0, stream>>>(feat, (size_t)BATCH * 512 * NPTS, -3.0e38f);

  dim3 ggrid(NPTS / 64, NPTS / 64, BATCH);
  int pblocks = NP_TOT / 16;   // 2560

  // ---- stage 1: knn(x, C=3) -> adaptive conv 1 -> feat rows [0,64)
  norms_k<<<cdiv(BATCH * NPTS, 128), 128, 0, stream>>>(x, 3 * NPTS, 3, xx);
  gram_k<<<ggrid, 128, 0, stream>>>(x, 3 * NPTS, 3, G);
  topk_k<<<cdiv(BATCH * NPTS, 128), 128, 0, stream>>>(G, xx, idx);
  adapt_conv1_k<<<pblocks, 128, 0, stream>>>(x, idx,
      ac1_w0, ac1_g0, ac1_b0, w1r1, ac1_g1, ac1_b1, feat, 0);

  // ---- stage 2: knn(x1, C=64) -> adaptive conv 2 -> feat rows [64,128)
  norms_k<<<cdiv(BATCH * NPTS, 128), 128, 0, stream>>>(feat, 512 * NPTS, 64, xx);
  gram_k<<<ggrid, 128, 0, stream>>>(feat, 512 * NPTS, 64, G);
  topk_k<<<cdiv(BATCH * NPTS, 128), 128, 0, stream>>>(G, xx, idx);
  adapt_conv2_k<<<pblocks, 128, 0, stream>>>(x, feat, 512 * NPTS, idx,
      ac2_w0, ac2_g0, ac2_b0, w1r2, ac2_g1, ac2_b1, feat, 64);

  // ---- stage 3: knn(x2, C=64) -> edge MLP c3 -> feat rows [128,256)
  const float* x2p = feat + 64 * NPTS;
  norms_k<<<cdiv(BATCH * NPTS, 128), 128, 0, stream>>>(x2p, 512 * NPTS, 64, xx);
  gram_k<<<ggrid, 128, 0, stream>>>(x2p, 512 * NPTS, 64, G);
  topk_k<<<cdiv(BATCH * NPTS, 128), 128, 0, stream>>>(G, xx, idx);
  graph_mlp_k<<<pblocks, 128, 0, stream>>>(x2p, 512 * NPTS, 64, idx,
      wc3, c3_g, c3_b, 128, feat, 128);

  // ---- stage 4: knn(x3, C=128) -> edge MLP c4 -> feat rows [256,512)
  const float* x3p = feat + 128 * NPTS;
  norms_k<<<cdiv(BATCH * NPTS, 128), 128, 0, stream>>>(x3p, 512 * NPTS, 128, xx);
  gram_k<<<ggrid, 128, 0, stream>>>(x3p, 512 * NPTS, 128, G);
  topk_k<<<cdiv(BATCH * NPTS, 128), 128, 0, stream>>>(G, xx, idx);
  graph_mlp_k<<<pblocks, 128, 0, stream>>>(x3p, 512 * NPTS, 128, idx,
      wc4, c4_g, c4_b, 256, feat, 256);

  // ---- c5 conv1d + pooling + FC head
  conv1d_c5_k<<<BATCH * NPTS / 16, 128, 0, stream>>>(feat, wc5, c5_g, c5_b, h5);
  pool_k<<<BATCH * 1024 / 4, 128, 0, stream>>>(h5, pooled);
  fc_k<<<cdiv(BATCH * 512, 4), 128, 0, stream>>>(pooled, 2048, h1_w, nullptr, h1_g, h1_b, 1, 512, fb1);
  fc_k<<<cdiv(BATCH * 256, 4), 128, 0, stream>>>(fb1, 512, h2_w, h2_bias, h2_g, h2_b, 1, 256, fb2);
  fc_k<<<cdiv(BATCH * 40, 4), 128, 0, stream>>>(fb2, 256, h3_w, h3_bias, nullptr, nullptr, 0, 40, out);
  (void)in_sizes; (void)n_in; (void)out_size; (void)ws_size;
}
